// GraphTemporalRNNConv_34780645163652
// MI455X (gfx1250) — compile-verified, matrix operand order checked by reading, not run on hardware
//
#include <hip/hip_runtime.h>
#include <math.h>

// Problem constants (from reference)
#define Nn 5000
#define Ee 320000
#define Dd 200          // in_dim == hid_dim
#define NB2 100         // bdd bases, 2x2 blocks
#define KP 224          // K padded to 7 x 32
#define NP 5008         // node rows padded to 313 x 16
#define WP 208          // weight rows (output cols) padded to 13 x 16

typedef __attribute__((ext_vector_type(16))) _Float16 v16h;
typedef __attribute__((ext_vector_type(8)))  _Float16 v8h;
typedef __attribute__((ext_vector_type(8)))  float    v8f;

// ---------------------------------------------------------------------------
// Packed-fragment loader.  Buffers are zero-padded f16, row-major with ld=KP.
// Per ISA 7.12.2 (16-bit A 16x32): lane<16 row M=lane, halfs 0-7 -> K=k0..k0+7,
// halfs 8-15 -> K=k0+16..k0+23; lanes 16-31 same rows with K offset +8.
// base already points at (row, koff) where koff = (lane&16)?8:0, so each
// fragment is exactly two aligned 16-byte loads (global_load_b128).
// ---------------------------------------------------------------------------
__device__ __forceinline__ v16h frag_packed(const _Float16* __restrict__ base, int k0) {
  v8h lo = *(const v8h*)(base + k0);
  v8h hi = *(const v8h*)(base + k0 + 16);
  return __builtin_shufflevector(lo, hi, 0, 1, 2, 3, 4, 5, 6, 7,
                                 8, 9, 10, 11, 12, 13, 14, 15);
}

// ---------------------------------------------------------------------------
// Generic f32 -> padded-f16 pack: dst[r*KP + c] = src[off + r*rstride + c*cstride]
// (zero outside rows x cols).  Handles row-major copy, transpose, and the
// stride-2 dynamic_temporal slices with one kernel.
// ---------------------------------------------------------------------------
__global__ void pack_f16_kernel(const float* __restrict__ src,
                                _Float16* __restrict__ dst,
                                int rows, int cols,
                                long long rstride, long long cstride,
                                long long off, int rpad) {
  long long t = (long long)blockIdx.x * blockDim.x + threadIdx.x;
  if (t >= (long long)rpad * KP) return;
  int r = (int)(t / KP);
  int c = (int)(t % KP);
  float v = (r < rows && c < cols) ? src[off + r * rstride + c * cstride] : 0.0f;
  dst[t] = (_Float16)v;
}

// ---------------------------------------------------------------------------
// Per-edge inverse inter-event-time norms (forward + reverse graphs)
// ---------------------------------------------------------------------------
__global__ void edge_norm_kernel(const float* __restrict__ edge_time,
                                 const float* __restrict__ nlet,   // (N, N+1, 2)
                                 const int* __restrict__ src,
                                 const int* __restrict__ dst,
                                 const int* __restrict__ nid,
                                 float* __restrict__ norm_f,
                                 float* __restrict__ norm_r) {
  int e = blockIdx.x * blockDim.x + threadIdx.x;
  if (e >= Ee) return;
  int gs = nid[src[e]];
  int gd = nid[dst[e]];
  float t = edge_time[e];
  float dtf = t - nlet[((long long)gd * (Nn + 1) + gs) * 2 + 0];
  float dtr = t - nlet[((long long)gs * (Nn + 1) + gd) * 2 + 1];
  norm_f[e] = fminf(1.0f / fmaxf(dtf, 1e-10f), 10.0f);
  norm_r[e] = fminf(1.0f / fmaxf(dtr, 1e-10f), 10.0f);
}

// ---------------------------------------------------------------------------
// RGCN-BDD message + scatter (f32, L2-resident gathers + atomics):
// msg[e,b,o] = sum_i h[gat[e],b,i] * W[rel,b,i,o] * norm[e]  -> agg[sct[e]]
// One thread handles 4 blocks (8 values); 25 threads per edge.
// ---------------------------------------------------------------------------
__global__ void edge_msg_kernel(const float* __restrict__ h,
                                const int* __restrict__ gatI,
                                const int* __restrict__ sctI,
                                const int* __restrict__ rel,
                                const float* __restrict__ norm,
                                const float* __restrict__ W,    // (R, NB2, 4)
                                float* __restrict__ agg) {
  const int GP = NB2 / 4;
  long long t = (long long)blockIdx.x * blockDim.x + threadIdx.x;
  if (t >= (long long)Ee * GP) return;
  int e = (int)(t / GP);
  int g = (int)(t % GP);
  int s = gatI[e], d = sctI[e], r = rel[e];
  float nm = norm[e];
  const float* hp = h + (long long)s * Dd + g * 8;
  const float* wp = W + ((long long)r * NB2 + g * 4) * 4;
  float* ap = agg + (long long)d * Dd + g * 8;
#pragma unroll
  for (int b = 0; b < 4; ++b) {
    float h0 = hp[b * 2 + 0], h1 = hp[b * 2 + 1];
    float m0 = (h0 * wp[b * 4 + 0] + h1 * wp[b * 4 + 2]) * nm;
    float m1 = (h0 * wp[b * 4 + 1] + h1 * wp[b * 4 + 3]) * nm;
    atomicAdd(&ap[b * 2 + 0], m0);
    atomicAdd(&ap[b * 2 + 1], m1);
  }
}

// ---------------------------------------------------------------------------
// Node update: out = act?(tanh):(id) of (agg + bias + A @ B), A packed [NP][KP],
// B pre-transposed/packed [WP][KP].  Writes f32 result and packed-f16 copy.
// One wave per 16x16 tile; branch-free inner loop: 4x b128 + 1x wmma per k-step.
// ---------------------------------------------------------------------------
__global__ void node_gemm_wmma(const _Float16* __restrict__ Ap,
                               const _Float16* __restrict__ BT,
                               const float* __restrict__ agg,
                               const float* __restrict__ bias,
                               float* __restrict__ outf,
                               _Float16* __restrict__ outp,
                               int act) {
  int lane = threadIdx.x;
  int ct = blockIdx.x * blockDim.y + threadIdx.y;   // column tile (0..12)
  int rt = blockIdx.y;                              // row tile
  if (ct * 16 >= Dd) return;                        // uniform per wave
  int koff = (lane & 16) ? 8 : 0;
  const _Float16* pA = Ap + (long long)(rt * 16 + (lane & 15)) * KP + koff;
  const _Float16* pB = BT + (long long)(ct * 16 + (lane & 15)) * KP + koff;

  v8f c = {};
#pragma unroll
  for (int k0 = 0; k0 < KP; k0 += 32) {
    v16h a = frag_packed(pA, k0);
    v16h b = frag_packed(pB, k0);
    c = __builtin_amdgcn_wmma_f32_16x16x32_f16(false, a, false, b,
                                               (short)0, c, false, false);
  }

  int ocol = ct * 16 + (lane & 15);
#pragma unroll
  for (int r = 0; r < 8; ++r) {
    int orow = rt * 16 + r + ((lane & 16) ? 8 : 0);
    bool in = (orow < Nn) && (ocol < Dd);
    float v = 0.0f;
    if (in) {
      v = c[r] + agg[(long long)orow * Dd + ocol] + bias[ocol];
      v = act ? tanhf(v) : v;
      outf[(long long)orow * Dd + ocol] = v;
    }
    outp[(long long)orow * KP + ocol] = (_Float16)(in ? v : 0.0f);
  }
}

// ---------------------------------------------------------------------------
// Fused RNN step: out[n,0,:,dir] = tanh(X @ Wih^T + H0 @ Whh^T + b_ih + b_hh)
// All four operands pre-packed; two wmma chains share one accumulator.
// ---------------------------------------------------------------------------
__global__ void rnn_gemm_wmma(const _Float16* __restrict__ Xp,
                              const _Float16* __restrict__ WihP,
                              const _Float16* __restrict__ H0p,
                              const _Float16* __restrict__ WhhP,
                              const float* __restrict__ b_ih,
                              const float* __restrict__ b_hh,
                              float* __restrict__ out, int dir) {
  int lane = threadIdx.x;
  int ct = blockIdx.x * blockDim.y + threadIdx.y;
  int rt = blockIdx.y;
  if (ct * 16 >= Dd) return;
  int koff = (lane & 16) ? 8 : 0;
  long long rowoff = (long long)(rt * 16 + (lane & 15)) * KP + koff;
  long long coloff = (long long)(ct * 16 + (lane & 15)) * KP + koff;
  const _Float16* pA1 = Xp + rowoff;
  const _Float16* pB1 = WihP + coloff;
  const _Float16* pA2 = H0p + rowoff;
  const _Float16* pB2 = WhhP + coloff;

  v8f c = {};
#pragma unroll
  for (int k0 = 0; k0 < KP; k0 += 32) {
    v16h a1 = frag_packed(pA1, k0);
    v16h b1 = frag_packed(pB1, k0);
    c = __builtin_amdgcn_wmma_f32_16x16x32_f16(false, a1, false, b1,
                                               (short)0, c, false, false);
    v16h a2 = frag_packed(pA2, k0);
    v16h b2 = frag_packed(pB2, k0);
    c = __builtin_amdgcn_wmma_f32_16x16x32_f16(false, a2, false, b2,
                                               (short)0, c, false, false);
  }

  int ocol = ct * 16 + (lane & 15);
  if (ocol >= Dd) return;
#pragma unroll
  for (int r = 0; r < 8; ++r) {
    int orow = rt * 16 + r + ((lane & 16) ? 8 : 0);
    if (orow < Nn) {
      float v = c[r] + b_ih[ocol] + b_hh[ocol];
      out[(long long)orow * (2 * Dd) + ocol * 2 + dir] = tanhf(v);
    }
  }
}

// ---------------------------------------------------------------------------
extern "C" void kernel_launch(void* const* d_in, const int* in_sizes, int n_in,
                              void* d_out, int out_size, void* d_ws, size_t ws_size,
                              hipStream_t stream) {
  (void)in_sizes; (void)n_in; (void)out_size; (void)ws_size;
  const float* static_temporal  = (const float*)d_in[0];
  const float* dynamic_temporal = (const float*)d_in[1];
  const float* nlet             = (const float*)d_in[2];
  const float* edge_time        = (const float*)d_in[3];
  const float* W1               = (const float*)d_in[4];
  const float* loop1            = (const float*)d_in[5];
  const float* b1               = (const float*)d_in[6];
  const float* W2               = (const float*)d_in[7];
  const float* loop2             = (const float*)d_in[8];
  const float* b2               = (const float*)d_in[9];
  const float* W_ih             = (const float*)d_in[10];
  const float* W_hh             = (const float*)d_in[11];
  const float* b_ih             = (const float*)d_in[12];
  const float* b_hh             = (const float*)d_in[13];
  const int*   src              = (const int*)d_in[14];
  const int*   dst              = (const int*)d_in[15];
  const int*   rel              = (const int*)d_in[16];
  const int*   nid              = (const int*)d_in[17];
  float* out = (float*)d_out;

  // ---- workspace layout ----
  // f32 region (element counts all multiples of 4 -> f16 region stays 16B aligned)
  float* wsf    = (float*)d_ws;
  float* norm_f = wsf;                               // Ee
  float* norm_r = norm_f + Ee;                       // Ee
  float* agg    = norm_r + Ee;                       // Nn*Dd
  float* h1f    = agg + (long long)Nn * Dd;          // Nn*Dd (f32 for edge gathers)
  // f16 region
  _Float16* wsp   = (_Float16*)(h1f + (long long)Nn * Dd);
  const long long ACT = (long long)NP * KP;          // 1,121,792 halfs
  const long long WGT = (long long)WP * KP;          // 46,592 halfs
  _Float16* embp   = wsp;               // packed static_temporal
  _Float16* dyn0p  = embp + ACT;        // packed dynamic_temporal[:,:,0]
  _Float16* dyn1p  = dyn0p + ACT;       // packed dynamic_temporal[:,:,1]
  _Float16* h1p    = dyn1p + ACT;       // packed layer-1 output
  _Float16* h2fp   = h1p + ACT;         // packed layer-2 forward output
  _Float16* h2rp   = h2fp + ACT;        // packed layer-2 reverse output
  _Float16* loop1T = h2rp + ACT;        // loop1 transposed-packed
  _Float16* loop2T = loop1T + WGT;
  _Float16* WihP   = loop2T + WGT;      // W_ih packed (already [n][k] for x@W^T)
  _Float16* WhhP   = WihP + WGT;

  const size_t aggBytes = (size_t)Nn * Dd * sizeof(float);
  const size_t actBytes = (size_t)ACT * sizeof(_Float16);
  int actPackBlocks = (int)((ACT + 255) / 256);
  int wgtPackBlocks = (int)((WGT + 255) / 256);

  // ---- 1) edge norms ----
  edge_norm_kernel<<<(Ee + 255) / 256, 256, 0, stream>>>(
      edge_time, nlet, src, dst, nid, norm_f, norm_r);

  // ---- 2) pack all GEMM operands to padded f16 ----
  pack_f16_kernel<<<actPackBlocks, 256, 0, stream>>>(static_temporal, embp,
      Nn, Dd, Dd, 1, 0, NP);
  pack_f16_kernel<<<actPackBlocks, 256, 0, stream>>>(dynamic_temporal, dyn0p,
      Nn, Dd, 2 * Dd, 2, 0, NP);
  pack_f16_kernel<<<actPackBlocks, 256, 0, stream>>>(dynamic_temporal, dyn1p,
      Nn, Dd, 2 * Dd, 2, 1, NP);
  // loopW used as B[k][n] -> pack transposed: dst[n][k] = loopW[k*Dd + n]
  pack_f16_kernel<<<wgtPackBlocks, 256, 0, stream>>>(loop1, loop1T,
      Dd, Dd, 1, Dd, 0, WP);
  pack_f16_kernel<<<wgtPackBlocks, 256, 0, stream>>>(loop2, loop2T,
      Dd, Dd, 1, Dd, 0, WP);
  // W_ih/W_hh used as x@W^T -> dst[n][k] = W[n*Dd + k] (straight copy + pad)
  pack_f16_kernel<<<wgtPackBlocks, 256, 0, stream>>>(W_ih, WihP,
      Dd, Dd, Dd, 1, 0, WP);
  pack_f16_kernel<<<wgtPackBlocks, 256, 0, stream>>>(W_hh, WhhP,
      Dd, Dd, Dd, 1, 0, WP);
  // zero GEMM-produced packed buffers (keeps K-pad columns 208..223 at zero)
  hipMemsetAsync(h1p, 0, actBytes, stream);
  hipMemsetAsync(h2fp, 0, actBytes, stream);
  hipMemsetAsync(h2rp, 0, actBytes, stream);

  dim3 gblk(32, 4);                         // 4 waves/block, one 16x16 tile each
  dim3 ggrd((13 + 3) / 4, (Nn + 15) / 16);
  long long msgThreads = (long long)Ee * (NB2 / 4);
  int msgBlocks = (int)((msgThreads + 255) / 256);

  // ---- 3) forward graph, layer 1 (tanh) ----
  hipMemsetAsync(agg, 0, aggBytes, stream);
  edge_msg_kernel<<<msgBlocks, 256, 0, stream>>>(static_temporal, src, dst, rel, norm_f, W1, agg);
  node_gemm_wmma<<<ggrd, gblk, 0, stream>>>(embp, loop1T, agg, b1, h1f, h1p, 1);

  // ---- 4) forward graph, layer 2 (linear) ----
  hipMemsetAsync(agg, 0, aggBytes, stream);
  edge_msg_kernel<<<msgBlocks, 256, 0, stream>>>(h1f, src, dst, rel, norm_f, W2, agg);
  node_gemm_wmma<<<ggrd, gblk, 0, stream>>>(h1p, loop2T, agg, b2, h1f, h2fp, 0);

  // ---- 5) reversed graph, layer 1 (tanh) ----
  hipMemsetAsync(agg, 0, aggBytes, stream);
  edge_msg_kernel<<<msgBlocks, 256, 0, stream>>>(static_temporal, dst, src, rel, norm_r, W1, agg);
  node_gemm_wmma<<<ggrd, gblk, 0, stream>>>(embp, loop1T, agg, b1, h1f, h1p, 1);

  // ---- 6) reversed graph, layer 2 (linear) ----
  hipMemsetAsync(agg, 0, aggBytes, stream);
  edge_msg_kernel<<<msgBlocks, 256, 0, stream>>>(h1f, dst, src, rel, norm_r, W2, agg);
  node_gemm_wmma<<<ggrd, gblk, 0, stream>>>(h1p, loop2T, agg, b2, h1f, h2rp, 0);

  // ---- 7) fused RNN steps, interleaved (N,1,D,2) output ----
  rnn_gemm_wmma<<<ggrd, gblk, 0, stream>>>(h2fp, WihP, dyn0p, WhhP, b_ih, b_hh, out, 0);
  rnn_gemm_wmma<<<ggrd, gblk, 0, stream>>>(h2rp, WihP, dyn1p, WhhP, b_ih, b_hh, out, 1);
}
